// LSTMDirectionModel_76708115906740
// MI455X (gfx1250) — matrix operand order repaired and checked by reading, not compiled, not run on hardware
//
#include <hip/hip_runtime.h>
#include <stdint.h>

// LSTM: B=4096, T=512, I=3, H=32, C=3 ; gates = 4H = 128 (PyTorch order i,f,g,o)
#define B_  4096
#define T_  512
#define I_  3
#define H_  32
#define C_  3

typedef __attribute__((ext_vector_type(16))) _Float16 v16h;
typedef __attribute__((ext_vector_type(8)))  float    v8f;
typedef __attribute__((ext_vector_type(2)))  float    v2f;

typedef union { v16h v; uint32_t u[8]; } frag16;

__device__ __forceinline__ float vtanh(float x) {
#if __has_builtin(__builtin_amdgcn_tanhf)
    return __builtin_amdgcn_tanhf(x);
#else
    float r;
    asm("v_tanh_f32 %0, %1" : "=v"(r) : "v"(x));
    return r;
#endif
}
__device__ __forceinline__ uint32_t pkh(float a, float b) {
    union { uint32_t u; _Float16 h[2]; } p;
    p.h[0] = (_Float16)a; p.h[1] = (_Float16)b;
    return p.u;
}

// 2 waves per block; block owns a 16-row batch tile.
// Wave w handles gate tiles {2g + w : g=0..3} = hidden columns w*16..w*16+15 of
// all four gates (i,f,g,o), so the c/h elementwise update is wave-local.
// i/f/o weights (and bias) are pre-scaled by 0.5 so sigmoid = fma(tanh(acc),0.5,0.5).
__global__ __launch_bounds__(64) void lstm_wmma_kernel(
    const float* __restrict__ x,     // (B,T,I)
    const float* __restrict__ W_ih,  // (4H,I)
    const float* __restrict__ W_hh,  // (4H,H)
    const float* __restrict__ b_ih,  // (4H)
    const float* __restrict__ b_hh,  // (4H)
    const float* __restrict__ W_fc,  // (C,H)
    const float* __restrict__ b_fc,  // (C)
    float* __restrict__ out)         // (B,C)
{
    // double-buffered h_t (16 rows x 32 hidden, f16): one sync per step
    __shared__ _Float16 hbuf[2][16][32];

    const int tid  = threadIdx.x;
    const int cw   = tid >> 5;          // wave id = hidden-column group (0,1)
    const int lane = tid & 31;
    const int lo   = lane & 15;
    const int hi   = lane >> 4;         // 0 or 1
    const int base = blockIdx.x * 16;   // batch tile base

    // ---- W_hh as 4 f16 B-fragments (32x16), one per gate g, columns cw*16..+15.
    // B k order used here (same as the A fragment): VGPR v, lanes 0-15: K=2v,2v+1 ; lanes 16-31: K=16+2v,17+2v.
    frag16 whhB[4];
    v2f    xiB[4];   // [W_ih ; bias] f32 B-fragments (4x16) for V_WMMA_F32_16X16X4_F32
    #pragma unroll
    for (int g = 0; g < 4; ++g) {
        const int gate  = g * 32 + cw * 16 + lo;   // global gate row
        const float scl = (g == 2) ? 1.0f : 0.5f;  // g-gate (tanh) unscaled
        #pragma unroll
        for (int v = 0; v < 8; ++v) {
            const int k0 = 2 * v + (hi ? 16 : 0);
            whhB[g].u[v] = pkh(scl * W_hh[gate * H_ + k0],
                               scl * W_hh[gate * H_ + k0 + 1]);
        }
        // xiB B layout: V0: lanes 0-15 K=0, lanes 16-31 K=2 ; V1: K=1 / K=3(bias)
        xiB[g].x = scl * (hi ? W_ih[gate * I_ + 2] : W_ih[gate * I_ + 0]);
        xiB[g].y = scl * (hi ? (b_ih[gate] + b_hh[gate]) : W_ih[gate * I_ + 1]);
    }

    // ---- init h = 0 (buffer 0), c = 0
    #pragma unroll
    for (int k = 0; k < 8; ++k) hbuf[0][lo][cw * 16 + hi * 8 + k] = (_Float16)0.0f;
    float cst[8];
    #pragma unroll
    for (int r = 0; r < 8; ++r) cst[r] = 0.0f;

    const float* xp = x + (size_t)(base + lo) * (T_ * I_);

    for (int t = 0; t < T_; ++t) {
        const int rb = t & 1;          // read buffer
        const int wb = rb ^ 1;         // write buffer
        __syncthreads();               // h stores from prev step visible to both waves

        // x A-fragment (f32 16x4): row lo = [x0, x1, x2, 1]
        // lanes 0-15 hold K=0,1 ; lanes 16-31 hold K=2,3
        v2f xa;
        if (hi == 0) { xa.x = xp[t * I_ + 0]; xa.y = xp[t * I_ + 1]; }
        else         { xa.x = xp[t * I_ + 2]; xa.y = 1.0f; }

        // h A-fragment (16x32 f16):
        // lane l (M=l&15): VGPR v: K = 2v + (hi?16:0)
        frag16 aF;
        #pragma unroll
        for (int v = 0; v < 8; ++v) {
            const int k0 = 2 * v + (hi ? 16 : 0);
            aF.u[v] = *(const uint32_t*)&hbuf[rb][lo][k0];
        }

        // gates = [x,1] @ [W_ih;b]^T  (f32 WMMA, K=4)  then += h @ W_hh^T (f16 WMMA)
        v8f acc[4];
        #pragma unroll
        for (int g = 0; g < 4; ++g) {
            v8f z = {};
            acc[g] = __builtin_amdgcn_wmma_f32_16x16x4_f32(
                false, xa, false, xiB[g], (short)0, z, false, false);
            acc[g] = __builtin_amdgcn_wmma_f32_16x16x32_f16(
                false, aF.v, false, whhB[g].v, (short)0, acc[g], false, false);
        }

        // elementwise update; C/D layout: VGPR r, lane: M = r + 8*hi, col = cw*16 + lo
        #pragma unroll
        for (int r = 0; r < 8; ++r) {
            const float ig = fmaf(vtanh(acc[0][r]), 0.5f, 0.5f);   // pre-scaled by 0.5
            const float fg = fmaf(vtanh(acc[1][r]), 0.5f, 0.5f);
            const float gg = vtanh(acc[2][r]);                     // unscaled
            const float og = fmaf(vtanh(acc[3][r]), 0.5f, 0.5f);
            const float cv = fg * cst[r] + ig * gg;
            cst[r] = cv;
            const float hv = og * vtanh(cv);
            hbuf[wb][r + (hi ? 8 : 0)][cw * 16 + lo] = (_Float16)hv;
        }
    }
    __syncthreads();

    // ---- final FC: out = h_last @ W_fc^T + b_fc ; wave 0, lanes 0..15 (one row each)
    // T_ = 512 even -> final h is in buffer 0
    if (cw == 0 && hi == 0) {
        float a0 = b_fc[0], a1 = b_fc[1], a2 = b_fc[2];
        #pragma unroll
        for (int j = 0; j < H_; ++j) {
            const float hv = (float)hbuf[T_ & 1][lo][j];
            a0 += hv * W_fc[0 * H_ + j];
            a1 += hv * W_fc[1 * H_ + j];
            a2 += hv * W_fc[2 * H_ + j];
        }
        float* op = out + (size_t)(base + lo) * C_;
        op[0] = a0; op[1] = a1; op[2] = a2;
    }
}

extern "C" void kernel_launch(void* const* d_in, const int* in_sizes, int n_in,
                              void* d_out, int out_size, void* d_ws, size_t ws_size,
                              hipStream_t stream) {
    (void)in_sizes; (void)n_in; (void)out_size; (void)d_ws; (void)ws_size;
    const float* x    = (const float*)d_in[0];
    const float* W_ih = (const float*)d_in[1];
    const float* W_hh = (const float*)d_in[2];
    const float* b_ih = (const float*)d_in[3];
    const float* b_hh = (const float*)d_in[4];
    const float* W_fc = (const float*)d_in[5];
    const float* b_fc = (const float*)d_in[6];
    float* out = (float*)d_out;

    dim3 grid(B_ / 16);   // 256 batch tiles; 2 waves each -> 512 waves
    dim3 block(64);
    lstm_wmma_kernel<<<grid, block, 0, stream>>>(x, W_ih, W_hh, b_ih, b_hh, W_fc, b_fc, out);
}
